// RnnController_66752381714443
// MI455X (gfx1250) — matrix operand order repaired and checked by reading, not compiled, not run on hardware
//
#include <hip/hip_runtime.h>
#include <hip/hip_bf16.h>
#include <math.h>

#define BB    8192
#define HH    100
#define NOPS  5
#define TT    512
#define KX    224      // padded K for concatenated [x | h] (7 x 32)
#define HOFF  112      // where h starts inside K
#define N4H   400      // 4*H gate width
#define ROWS  16       // batch rows per workgroup
#define NTHR  256      // 8 waves

typedef __bf16 bf16;
typedef __attribute__((ext_vector_type(16))) bf16  v16bf;
typedef __attribute__((ext_vector_type(8)))  bf16  v8bf;
typedef __attribute__((ext_vector_type(8)))  float v8f;

__device__ __forceinline__ bf16 f2bf(float f) {
  unsigned u = __builtin_bit_cast(unsigned, f);
  u += 0x7fffu + ((u >> 16) & 1u);            // round-to-nearest-even
  unsigned short h = (unsigned short)(u >> 16);
  return __builtin_bit_cast(bf16, h);
}

__device__ __forceinline__ float sigm(float x) { return 1.0f / (1.0f + __expf(-x)); }

__global__ __launch_bounds__(NTHR, 1)
void rnn_controller_kernel(const float* __restrict__ x0,
                           const float* __restrict__ Wx,    // [H, 4H]
                           const float* __restrict__ Wh,    // [H, 4H]
                           const float* __restrict__ Wop,   // [H, NOPS]
                           const float* __restrict__ emb,   // [NOPS+1, H]
                           const float* __restrict__ u,     // [T, B, NOPS]
                           float* __restrict__ out)         // [B] lp, [B] ent, [T*B] ops(int32)
{
  // LDS-resident weights (bf16, transposed: row = output col n, K contiguous) + state
  __shared__ __attribute__((aligned(64))) bf16  sWt[N4H][KX];     // 179200 B
  __shared__ __attribute__((aligned(64))) bf16  sA[ROWS][KX];     //   7168 B  [x|h] bf16
  __shared__ __attribute__((aligned(64))) float sZ[ROWS][N4H];    //  25600 B  gate pre-acts
  __shared__ float sC[ROWS][HH];                                  //   6400 B
  __shared__ float sH[ROWS][HH];                                  //   6400 B
  __shared__ float sWo[HH * NOPS];                                //   2000 B
  __shared__ __attribute__((aligned(64))) bf16  sE[NOPS + 1][HOFF]; // 1344 B
  __shared__ float sL[ROWS][NOPS];
  __shared__ int   sOp[ROWS];

  const int tid  = threadIdx.x;
  const int base = blockIdx.x * ROWS;
  const int lane = tid & 31;
  const int wave = tid >> 5;
  const int half = lane >> 4;     // lane group 0/1
  const int l16  = lane & 15;

  // ---------- one-time setup ----------
  for (int i = tid; i < N4H * KX; i += NTHR) {
    int n = i / KX, k = i % KX;
    float w = 0.f;
    if (k < HH)                      w = Wx[k * N4H + n];
    else if (k >= HOFF && k < HOFF + HH) w = Wh[(k - HOFF) * N4H + n];
    sWt[n][k] = f2bf(w);
  }
  for (int i = tid; i < ROWS * KX; i += NTHR) {
    int r = i / KX, k = i % KX;
    float v = (k < HH) ? x0[(size_t)(base + r) * HH + k] : 0.f;   // h0 = 0
    sA[r][k] = f2bf(v);
  }
  for (int i = tid; i < ROWS * HH; i += NTHR) sC[i / HH][i % HH] = 0.f;
  for (int i = tid; i < HH * NOPS; i += NTHR) sWo[i] = Wop[i];
  for (int i = tid; i < (NOPS + 1) * HOFF; i += NTHR) {
    int r = i / HOFF, k = i % HOFF;
    sE[r][k] = f2bf(k < HH ? emb[r * HH + k] : 0.f);
  }
  __syncthreads();

  float lp_acc = 0.f, en_acc = 0.f;

  for (int t = 0; t < TT; ++t) {
    // ---- 1) z[16 x 400] = [x|h](16 x 224) @ W^T  via bf16 WMMA ----
    v16bf af[7];
    {
      const int k0 = half * 16;                  // K sub-offset: contiguous order, matches B
      #pragma unroll
      for (int ks = 0; ks < 7; ++ks) {
        v8bf lo = *(const v8bf*)&sA[l16][ks * 32 + k0];
        v8bf hi = *(const v8bf*)&sA[l16][ks * 32 + k0 + 8];
        v16bf a;
        #pragma unroll
        for (int e = 0; e < 8; ++e) { a[e] = lo[e]; a[8 + e] = hi[e]; }
        af[ks] = a;
      }
    }
    for (int nt = wave; nt < 25; nt += 8) {      // 25 N-tiles over 8 waves
      const int ncol = nt * 16 + l16;
      const int kb   = half * 16;                // B: lanes 0-15 K=0..15, 16-31 K=16..31
      v8f acc = {0.f, 0.f, 0.f, 0.f, 0.f, 0.f, 0.f, 0.f};
      #pragma unroll
      for (int ks = 0; ks < 7; ++ks) {
        v16bf b = *(const v16bf*)&sWt[ncol][ks * 32 + kb];
        acc = __builtin_amdgcn_wmma_f32_16x16x32_bf16(
                  false, af[ks], false, b, (short)0, acc, false, false);
      }
      const int m0 = half * 8;                   // C/D layout: lane half selects M 0-7 / 8-15
      #pragma unroll
      for (int v = 0; v < 8; ++v) sZ[m0 + v][ncol] = acc[v];
    }
    __syncthreads();

    // ---- 2) LSTM gates (i,f,c,o), new c/h ----
    for (int i = tid; i < ROWS * HH; i += NTHR) {
      int r = i / HH, j = i % HH;
      float zi = sZ[r][j], zf = sZ[r][j + HH], zc = sZ[r][j + 2 * HH], zo = sZ[r][j + 3 * HH];
      float cn = sigm(zf) * sC[r][j] + sigm(zi) * tanhf(zc);
      float hn = sigm(zo) * tanhf(cn);
      sC[r][j] = cn;
      sH[r][j] = hn;
      sA[r][HOFF + j] = f2bf(hn);                // h-half of next-step A
    }
    __syncthreads();

    // ---- 3) logits = 1.5 * tanh(h @ W_ops) ----
    if (tid < ROWS * NOPS) {
      int r = tid / NOPS, o = tid % NOPS;
      float d = 0.f;
      #pragma unroll 4
      for (int j = 0; j < HH; ++j) d += sH[r][j] * sWo[j * NOPS + o];
      sL[r][o] = 1.5f * tanhf(d);
    }
    __syncthreads();

    // ---- 4) Gumbel-max sample, xent, entropy ----
    if (tid < ROWS) {
      const int r = tid, grow = base + r;
      const float* urow = u + ((size_t)t * BB + grow) * NOPS;
      float l[NOPS], mx = -1e30f, best = -1e30f;
      int op = 0;
      #pragma unroll
      for (int o = 0; o < NOPS; ++o) { l[o] = sL[r][o]; mx = fmaxf(mx, l[o]); }
      #pragma unroll
      for (int o = 0; o < NOPS; ++o) {
        float g = -__logf(-__logf(urow[o] + 1e-9f) + 1e-9f);
        float s = l[o] + g;
        if (s > best) { best = s; op = o; }
      }
      float se = 0.f;
      #pragma unroll
      for (int o = 0; o < NOPS; ++o) se += __expf(l[o] - mx);
      float lse = mx + __logf(se);
      float cur = lse - l[op];
      lp_acc += cur;
      en_acc += cur * __expf(-cur);
      sOp[r] = op;
      ((int*)out)[2 * BB + (size_t)t * BB + grow] = op;
    }
    __syncthreads();

    // ---- 5) next x = emb[op] ----
    for (int i = tid; i < ROWS * HOFF; i += NTHR) {
      int r = i / HOFF, k = i % HOFF;
      sA[r][k] = sE[sOp[r]][k];                  // pads already zero in sE
    }
    __syncthreads();
  }

  if (tid < ROWS) {
    const int grow = base + tid;
    out[grow]      = lp_acc;
    out[BB + grow] = en_acc;
  }
}

extern "C" void kernel_launch(void* const* d_in, const int* in_sizes, int n_in,
                              void* d_out, int out_size, void* d_ws, size_t ws_size,
                              hipStream_t stream) {
  (void)in_sizes; (void)n_in; (void)d_ws; (void)ws_size; (void)out_size;
  const float* x0  = (const float*)d_in[0];
  const float* Wx  = (const float*)d_in[1];
  const float* Wh  = (const float*)d_in[2];
  const float* Wop = (const float*)d_in[3];
  const float* emb = (const float*)d_in[4];
  const float* u   = (const float*)d_in[5];
  float* out = (float*)d_out;
  dim3 grid(BB / ROWS), block(NTHR);
  rnn_controller_kernel<<<grid, block, 0, stream>>>(x0, Wx, Wh, Wop, emb, u, out);
}
